// Actor_90297392431551
// MI455X (gfx1250) — compile-verified, matrix operand order checked by reading, not run on hardware
//
#include <hip/hip_runtime.h>
#include <hip/hip_bf16.h>

#define Bn 131072
#define Cn 512
#define Jn 7
#define Hn 512
#define KD 526      // C + 2J
#define KP 544      // padded K = 17*32
#define MT 32       // rows per workgroup

typedef __attribute__((ext_vector_type(16))) __bf16        v16bf;
typedef __attribute__((ext_vector_type(8)))  float         v8f;
typedef __attribute__((ext_vector_type(4)))  unsigned int  u32x4;

union ABu { v16bf v; u32x4 u[2]; };

__device__ __forceinline__ unsigned short f2bf(float f) {
  unsigned int u = __float_as_uint(f);
  return (unsigned short)((u + 0x7FFFu + ((u >> 16) & 1u)) >> 16);  // RNE
}
__device__ __forceinline__ float bf2f(unsigned short h) {
  return __uint_as_float(((unsigned int)h) << 16);
}
__device__ __forceinline__ float selu_f(float x) {
  const float a = 1.6732632423543772f, s = 1.0507009873554805f;
  return x > 0.f ? s * x : s * a * (expf(x) - 1.f);
}

// ---- prep: W1 (526x512 f32, row-major) -> W1T (512x544 bf16, N-major), zero mask
__global__ void prep_w1t(const float* __restrict__ W1,
                         unsigned short* __restrict__ w1t,
                         int* __restrict__ mask) {
  int idx = blockIdx.x * blockDim.x + threadIdx.x;
  if (idx < Hn * KP) {
    int n = idx / KP, k = idx - n * KP;
    float v = (k < KD) ? W1[(size_t)k * Hn + n] : 0.f;
    w1t[idx] = f2bf(v);
  }
  if (idx < Bn) mask[idx] = 0;
}

// ---- prep: row-presence mask from coords (dup writes of identical value are fine)
__global__ void prep_mask(const int* __restrict__ coords, int* __restrict__ mask) {
  int idx = blockIdx.x * blockDim.x + threadIdx.x;
  if (idx < Bn) mask[coords[idx]] = 1;
}

__global__ __launch_bounds__(256) void actor_main(
    const float* __restrict__ features, const float* __restrict__ jnt_pos,
    const float* __restrict__ jnt_goal, const float* __restrict__ ln1_g,
    const float* __restrict__ ln1_b,    const float* __restrict__ b1,
    const float* __restrict__ ln2_g,    const float* __restrict__ ln2_b,
    const float* __restrict__ W2,       const float* __restrict__ b2,
    const unsigned short* __restrict__ w1t, const int* __restrict__ mask,
    float* __restrict__ out) {
  __shared__ __align__(16) unsigned short sA[MT * KP];  // 34816 B, reused for h

  const int lane = threadIdx.x & 31;
  const int w    = threadIdx.x >> 5;
  const int base = blockIdx.x * MT;

  // ---------- Phase 1: mask + LayerNorm1 + SELU -> bf16 A-tile in LDS ----------
  {
    float g16[16], bb16[16];
#pragma unroll
    for (int e = 0; e < 16; ++e) {           // per-column params, shared by all 4 rows
      int c = lane * 16 + e;
      g16[e]  = ln1_g[c];
      bb16[e] = ln1_b[c];
    }
    for (int i = 0; i < 4; ++i) {
      int r  = w * 4 + i;
      int gr = base + r;
      float mk = mask[gr] ? 1.f : 0.f;
      const float* frow = features + (size_t)gr * Cn + lane * 16;
      float x[16];
#pragma unroll
      for (int e = 0; e < 16; ++e) x[e] = frow[e] * mk;
      float s = 0.f, q = 0.f;
#pragma unroll
      for (int e = 0; e < 16; ++e) { s += x[e]; q += x[e] * x[e]; }
#pragma unroll
      for (int o = 16; o > 0; o >>= 1) { s += __shfl_xor(s, o, 32); q += __shfl_xor(q, o, 32); }
      float mean = s * (1.f / Cn);
      float rstd = rsqrtf(q * (1.f / Cn) - mean * mean + 1e-5f);
      unsigned short* arow = sA + r * KP;
#pragma unroll
      for (int e = 0; e < 16; ++e) {
        float xn = (x[e] - mean) * rstd * g16[e] + bb16[e];
        arow[14 + lane * 16 + e] = f2bf(selu_f(xn));
      }
      if (lane < 7)        arow[lane] = f2bf(jnt_pos[gr * Jn + lane]);
      else if (lane < 14)  arow[lane] = f2bf(jnt_goal[gr * Jn + lane - 7]);
      else                 arow[512 + lane] = 0;  // cols 526..543 zero pad
    }
  }
  __syncthreads();

  // ---------- Phase 2: GEMM1 via v_wmma_f32_16x16x32_bf16 ----------
  const int khalf = lane >> 4;   // 16-bit A/B layout: lane half selects K sub-block
  const int mlow  = lane & 15;
  v8f zero = {0.f, 0.f, 0.f, 0.f, 0.f, 0.f, 0.f, 0.f};
  v8f acc[2][4];
#pragma unroll
  for (int a = 0; a < 2; ++a)
#pragma unroll
    for (int b = 0; b < 4; ++b) acc[a][b] = zero;

  for (int kb = 0; kb < KP / 32; ++kb) {
    int kbase = kb * 32 + khalf * 8;
    ABu a0, a1;
    a0.u[0] = *(const u32x4*)(sA + mlow * KP + kbase);
    a0.u[1] = *(const u32x4*)(sA + mlow * KP + kbase + 16);
    a1.u[0] = *(const u32x4*)(sA + (16 + mlow) * KP + kbase);
    a1.u[1] = *(const u32x4*)(sA + (16 + mlow) * KP + kbase + 16);
#pragma unroll
    for (int nt = 0; nt < 4; ++nt) {
      int n = w * 64 + nt * 16 + mlow;
      const unsigned short* bp = w1t + (size_t)n * KP + kbase;
      ABu bm;
      bm.u[0] = *(const u32x4*)(bp);
      bm.u[1] = *(const u32x4*)(bp + 16);
      acc[0][nt] = __builtin_amdgcn_wmma_f32_16x16x32_bf16(
          false, a0.v, false, bm.v, (short)0, acc[0][nt], false, false);
      acc[1][nt] = __builtin_amdgcn_wmma_f32_16x16x32_bf16(
          false, a1.v, false, bm.v, (short)0, acc[1][nt], false, false);
    }
  }
  __syncthreads();  // A-tile no longer needed; reuse LDS for h (bf16 [32][512])

  unsigned short* sH = sA;
#pragma unroll
  for (int mt = 0; mt < 2; ++mt)
#pragma unroll
    for (int nt = 0; nt < 4; ++nt) {
      int col = w * 64 + nt * 16 + mlow;
      float b1v = b1[col];
#pragma unroll
      for (int v = 0; v < 8; ++v) {
        int row = mt * 16 + 8 * khalf + v;  // C/D layout: lanes16-31 hold M+8
        sH[row * Hn + col] = f2bf(acc[mt][nt][v] + b1v);
      }
    }
  __syncthreads();

  // ---------- Phase 3: LayerNorm2 + SELU + GEMM2(512x7) + tanh ----------
  {
    float g16[16], bb16[16];
#pragma unroll
    for (int e = 0; e < 16; ++e) {           // per-column params, shared by all 4 rows
      int c = lane * 16 + e;
      g16[e]  = ln2_g[c];
      bb16[e] = ln2_b[c];
    }
    float hv[4][16];
#pragma unroll
    for (int i = 0; i < 4; ++i) {
      int r = w * 4 + i;
      const unsigned short* hrow = sH + r * Hn + lane * 16;
#pragma unroll
      for (int e = 0; e < 16; ++e) hv[i][e] = bf2f(hrow[e]);
      float s = 0.f, q = 0.f;
#pragma unroll
      for (int e = 0; e < 16; ++e) { s += hv[i][e]; q += hv[i][e] * hv[i][e]; }
#pragma unroll
      for (int o = 16; o > 0; o >>= 1) { s += __shfl_xor(s, o, 32); q += __shfl_xor(q, o, 32); }
      float mean = s * (1.f / Hn);
      float rstd = rsqrtf(q * (1.f / Hn) - mean * mean + 1e-5f);
#pragma unroll
      for (int e = 0; e < 16; ++e)
        hv[i][e] = selu_f((hv[i][e] - mean) * rstd * g16[e] + bb16[e]);
    }
    // GEMM2: each W2 row load is shared across the wave's 4 output rows
    float oj[4][7];
#pragma unroll
    for (int i = 0; i < 4; ++i)
#pragma unroll
      for (int j = 0; j < 7; ++j) oj[i][j] = 0.f;
#pragma unroll
    for (int e = 0; e < 16; ++e) {
      const float* wr = W2 + (size_t)(lane * 16 + e) * Jn;
      float w2v[7];
#pragma unroll
      for (int j = 0; j < 7; ++j) w2v[j] = wr[j];
#pragma unroll
      for (int i = 0; i < 4; ++i)
#pragma unroll
        for (int j = 0; j < 7; ++j) oj[i][j] += hv[i][e] * w2v[j];
    }
#pragma unroll
    for (int i = 0; i < 4; ++i) {
#pragma unroll
      for (int j = 0; j < 7; ++j)
#pragma unroll
        for (int o = 16; o > 0; o >>= 1) oj[i][j] += __shfl_xor(oj[i][j], o, 32);
      if (lane == 0) {
        int gr = base + w * 4 + i;
#pragma unroll
        for (int j = 0; j < 7; ++j) out[(size_t)gr * Jn + j] = tanhf(oj[i][j] + b2[j]);
      }
    }
  }
}

extern "C" void kernel_launch(void* const* d_in, const int* in_sizes, int n_in,
                              void* d_out, int out_size, void* d_ws, size_t ws_size,
                              hipStream_t stream) {
  const float* features = (const float*)d_in[0];
  const int*   coords   = (const int*)d_in[1];
  const float* jnt_pos  = (const float*)d_in[2];
  const float* jnt_goal = (const float*)d_in[3];
  const float* ln1_g    = (const float*)d_in[4];
  const float* ln1_b    = (const float*)d_in[5];
  const float* W1       = (const float*)d_in[6];
  const float* b1       = (const float*)d_in[7];
  const float* ln2_g    = (const float*)d_in[8];
  const float* ln2_b    = (const float*)d_in[9];
  const float* W2       = (const float*)d_in[10];
  const float* b2       = (const float*)d_in[11];

  unsigned short* w1t = (unsigned short*)d_ws;                 // 512*544*2 = 557056 B
  int* mask = (int*)((char*)d_ws + (size_t)Hn * KP * 2);       // +131072*4 B
  float* out = (float*)d_out;

  prep_w1t<<<(Hn * KP + 255) / 256, 256, 0, stream>>>(W1, w1t, mask);
  prep_mask<<<(Bn + 255) / 256, 256, 0, stream>>>(coords, mask);
  actor_main<<<Bn / MT, 256, 0, stream>>>(features, jnt_pos, jnt_goal, ln1_g, ln1_b,
                                          b1, ln2_g, ln2_b, W2, b2, w1t, mask, out);
}
